// VimBlock_6511170421678
// MI455X (gfx1250) — compile-verified, matrix operand order checked by reading, not run on hardware
//
#include <hip/hip_runtime.h>
#include <hip/hip_bf16.h>
#include <math.h>

// ---------------- problem constants (from reference) ----------------
#define DIMX    512
#define DSTATE  16
#define DCONVX  4
#define DINNERX 1024
#define DTRANKX 32
#define BBX     2
#define LLX     2048
#define MROWS   (BBX * LLX)             // 4096 token rows
#define XPROJN  (DTRANKX + 2 * DSTATE)  // 64

typedef __attribute__((ext_vector_type(16))) _Float16 v16h;
typedef __attribute__((ext_vector_type(8)))  _Float16 v8h;
typedef __attribute__((ext_vector_type(8)))  float    v8f;
typedef unsigned int u32x4 __attribute__((ext_vector_type(4)));
typedef int          i32x4 __attribute__((ext_vector_type(4)));
typedef int          i32x8 __attribute__((ext_vector_type(8)));

union FragAB { v16h v; v8h h[2]; };

// --------- Tensor Data Mover availability (probe-verified builtins) ---------
#if defined(__has_builtin)
#  if __has_builtin(__builtin_amdgcn_tensor_load_to_lds) && __has_builtin(__builtin_amdgcn_s_wait_tensorcnt)
#    define HAVE_TDM 1
#  endif
#endif
#ifndef HAVE_TDM
#  define HAVE_TDM 0
#endif

#if HAVE_TDM
// LDS byte offset of a __shared__ object (flat -> addrspace(3) strips aperture)
__device__ __forceinline__ unsigned ldsOffset(const void* p) {
    return (unsigned)(unsigned long long)(__attribute__((address_space(3))) const char*)(p);
}

// TDM 2-D f16 tile load: tileRows x 32 elements, row pitch strideElems,
// LDS pad 16 B after every 64 B -> matches the 40-half padded LDS pitch.
// D# built per CDNA5 ISA ch.8: group0 = {flags, lds_addr, global_addr, type=2},
// group1 = {data_size=2B, pad(16DW,+4DW), dims, tile dims, dim0 stride}.
__device__ __forceinline__ void tdm_load_2d_f16(unsigned ldsOff, const _Float16* g,
                                                int tileRows, long long strideElems)
{
    unsigned long long ga = (unsigned long long)(const void*)g;
    u32x4 g0;
    g0[0] = 1u;                                              // count=1, user D#
    g0[1] = ldsOff;                                          // lds_addr
    g0[2] = (unsigned)ga;                                    // global_addr[31:0]
    g0[3] = (unsigned)((ga >> 32) & 0x1ffffffULL) | (2u << 30); // [56:32], type=2
    unsigned long long s0 = (unsigned long long)strideElems;
    unsigned dim0 = (unsigned)s0;                            // tensor_dim0 (elems)
    unsigned dim1 = (unsigned)tileRows;                      // tensor_dim1
    i32x8 g1;
    g1[0] = (int)((1u << 16)          // data_size = 2 bytes
                | (1u << 20)          // pad_enable
                | (3u << 22)          // pad_interval: 16 DWORDs (64 B)
                | (3u << 25));        // pad_amount:   4 DWORDs (16 B)
    g1[1] = (int)((dim0 & 0xffffu) << 16);
    g1[2] = (int)(((dim0 >> 16) & 0xffffu) | ((dim1 & 0xffffu) << 16));
    g1[3] = (int)(((dim1 >> 16) & 0xffffu) | (32u << 16));   // tile_dim0 = 32
    g1[4] = (int)((unsigned)tileRows & 0xffffu);             // tile_dim1, tile_dim2=0
    g1[5] = (int)(s0 & 0xffffffffULL);                       // dim0_stride[31:0]
    g1[6] = (int)((s0 >> 32) & 0xffffULL);                   // dim0_stride[47:32]
    g1[7] = 0;
    i32x4 gz = {0, 0, 0, 0};
#if __clang_major__ >= 23
    i32x8 gz8 = {0, 0, 0, 0, 0, 0, 0, 0};
    __builtin_amdgcn_tensor_load_to_lds(g0, g1, gz, gz, gz8, 0);
#else
    __builtin_amdgcn_tensor_load_to_lds(g0, g1, gz, gz, 0);
#endif
}
#endif // HAVE_TDM

// ---------------------------------------------------------------
// LayerNorm: one wave per row of 512; writes f16 xn and time-flipped xn.
// ---------------------------------------------------------------
__global__ __launch_bounds__(256)
void layernorm_kernel(const float* __restrict__ x, const float* __restrict__ g,
                      const float* __restrict__ b,
                      _Float16* __restrict__ xnh, _Float16* __restrict__ xnrh)
{
    const int wave = threadIdx.x >> 5;
    const int lane = threadIdx.x & 31;
    const int row  = blockIdx.x * 8 + wave;
    const int bb   = row >> 11;                       // row / LLX
    const int l    = row & (LLX - 1);
    const int rrow = bb * LLX + (LLX - 1 - l);
    const float* xr = x + (size_t)row * DIMX;

    float v[16];
    float s = 0.0f;
#pragma unroll
    for (int i = 0; i < 16; ++i) { v[i] = xr[lane + i * 32]; s += v[i]; }
#pragma unroll
    for (int o = 16; o > 0; o >>= 1) s += __shfl_xor(s, o, 32);
    const float mu = s * (1.0f / DIMX);

    float q = 0.0f;
#pragma unroll
    for (int i = 0; i < 16; ++i) { float d = v[i] - mu; q += d * d; }
#pragma unroll
    for (int o = 16; o > 0; o >>= 1) q += __shfl_xor(q, o, 32);
    const float inv = rsqrtf(q * (1.0f / DIMX) + 1e-5f);

#pragma unroll
    for (int i = 0; i < 16; ++i) {
        const int c = lane + i * 32;
        const _Float16 r = (_Float16)((v[i] - mu) * inv * g[c] + b[c]);
        xnh [(size_t)row  * DIMX + c] = r;
        xnrh[(size_t)rrow * DIMX + c] = r;
    }
}

// ---------------------------------------------------------------
// Weight prep: W[K][N] fp32 -> Wt[N][K] f16 (transpose + convert, once).
// ---------------------------------------------------------------
__global__ __launch_bounds__(256)
void wtrans_kernel(const float* __restrict__ W, _Float16* __restrict__ Wt,
                   int K, int N)
{
    const int i = blockIdx.x * 256 + threadIdx.x;
    if (i < K * N) {
        const int k = i / N, n = i - k * N;
        Wt[(size_t)n * K + k] = (_Float16)W[i];
    }
}

// ---------------------------------------------------------------
// WMMA GEMM: f16 A[M][lda] row-major, f16 Bt[N][K] (pre-transposed),
// fp32 C out (+ optional f16 copy Ch). Block tile 64 x BN, K-step 32,
// 8 waves as 2(M) x 4(N); wave owns 32 x (BN/4): BN=128 -> 4 wmma.
// Double-buffered LDS; tiles staged by the Tensor Data Mover (wave 0,
// TENSORcnt) when available, else cooperative b128 copies.
// ---------------------------------------------------------------
template <int BN>
__global__ __launch_bounds__(256)
void gemm_wmma(const _Float16* __restrict__ A, const _Float16* __restrict__ Bt,
               float* __restrict__ C, int M, int N, int K,
               int lda, int ldc,
               const float* __restrict__ bias,
               const float* __restrict__ residual,
               _Float16* __restrict__ Ch)
{
    constexpr int SN = BN / 64;   // 16-wide N-subtiles per wave

    __shared__ __align__(16) _Float16 sA[2][64][40];   // [buf][m][k], 80 B pitch
    __shared__ __align__(16) _Float16 sB[2][BN][40];   // [buf][n][k]

    const int m0   = blockIdx.x * 64;
    const int n0   = blockIdx.y * BN;
    const int tid  = threadIdx.x;
    const int lane = tid & 31;
    const int wave = tid >> 5;
    const int waveM = wave >> 2;   // 0..1
    const int waveN = wave & 3;    // 0..3
    const int NK = K >> 5;

#if !HAVE_TDM
    auto coopStage = [&](int kt, int buf) {
        const int r = tid >> 2, cg = (tid & 3) * 8;
        *(v8h*)&sA[buf][r][cg] = *(const v8h*)(A + (size_t)(m0 + r) * lda + kt * 32 + cg);
#pragma unroll
        for (int c = 0; c < SN; ++c) {
            const int ci  = tid + 256 * c;
            const int row = ci >> 2, bg = (ci & 3) * 8;
            *(v8h*)&sB[buf][row][bg] = *(const v8h*)(Bt + (size_t)(n0 + row) * K + kt * 32 + bg);
        }
    };
#endif

    v8f acc[2][SN];
#pragma unroll
    for (int s = 0; s < 2; ++s)
#pragma unroll
        for (int t = 0; t < SN; ++t)
#pragma unroll
            for (int r = 0; r < 8; ++r) acc[s][t][r] = 0.0f;

    // stage tile 0
#if HAVE_TDM
    if (tid < 32) {
        tdm_load_2d_f16(ldsOffset(&sA[0][0][0]), A + (size_t)m0 * lda, 64, lda);
        tdm_load_2d_f16(ldsOffset(&sB[0][0][0]), Bt + (size_t)n0 * K, BN, K);
        __builtin_amdgcn_s_wait_tensorcnt(0);
    }
#else
    coopStage(0, 0);
#endif
    __syncthreads();

    for (int kt = 0; kt < NK; ++kt) {
        const int cur = kt & 1;

        // start async fill of the other buffer before computing
#if HAVE_TDM
        if (kt + 1 < NK && tid < 32) {
            tdm_load_2d_f16(ldsOffset(&sA[1 - cur][0][0]),
                            A + (size_t)m0 * lda + (kt + 1) * 32, 64, lda);
            tdm_load_2d_f16(ldsOffset(&sB[1 - cur][0][0]),
                            Bt + (size_t)n0 * K + (kt + 1) * 32, BN, K);
        }
#else
        if (kt + 1 < NK) coopStage(kt + 1, 1 - cur);
#endif

        // fragment loads per CDNA5 16-bit A/B VGPR layout:
        // lanes 0-15: K[0..7] in v0..3, K[16..23] in v4..7 ; lanes 16-31: +8
        const int hr = lane & 15;
        const int kb = (lane >> 4) * 8;
        FragAB fbf[SN];
#pragma unroll
        for (int t = 0; t < SN; ++t) {
            fbf[t].h[0] = *(const v8h*)&sB[cur][waveN * (16 * SN) + t * 16 + hr][kb];
            fbf[t].h[1] = *(const v8h*)&sB[cur][waveN * (16 * SN) + t * 16 + hr][kb + 16];
        }
#pragma unroll
        for (int s = 0; s < 2; ++s) {
            FragAB fa;
            fa.h[0] = *(const v8h*)&sA[cur][waveM * 32 + s * 16 + hr][kb];
            fa.h[1] = *(const v8h*)&sA[cur][waveM * 32 + s * 16 + hr][kb + 16];
#pragma unroll
            for (int t = 0; t < SN; ++t)
                acc[s][t] = __builtin_amdgcn_wmma_f32_16x16x32_f16(
                    false, fa.v, false, fbf[t].v, (short)0, acc[s][t], false, false);
        }

#if HAVE_TDM
        if (kt + 1 < NK && tid < 32) __builtin_amdgcn_s_wait_tensorcnt(0);
#endif
        __syncthreads();
    }

    // epilogue: C/D layout: lane<16 -> N=lane, M=r ; lane>=16 -> N=lane-16, M=r+8
#pragma unroll
    for (int t = 0; t < SN; ++t) {
        const int nIdx = n0 + waveN * (16 * SN) + t * 16 + (lane & 15);
        const float bv = bias ? bias[nIdx] : 0.0f;
#pragma unroll
        for (int s = 0; s < 2; ++s) {
            const int mBase = m0 + waveM * 32 + s * 16 + (lane >> 4) * 8;
#pragma unroll
            for (int r = 0; r < 8; ++r) {
                const int mm = mBase + r;
                float v = acc[s][t][r] + bv;
                if (residual) v += residual[(size_t)mm * ldc + nIdx];
                C[(size_t)mm * ldc + nIdx] = v;
                if (Ch) Ch[(size_t)mm * ldc + nIdx] = (_Float16)v;
            }
        }
    }
}

// ---------------------------------------------------------------
// Depthwise causal conv (DCONV=4) + SiLU; writes fp32 (scan) + f16 (GEMM).
// ---------------------------------------------------------------
__global__ __launch_bounds__(256)
void conv_silu_kernel(const float* __restrict__ xz, const float* __restrict__ cw,
                      const float* __restrict__ cb,
                      float* __restrict__ xic, _Float16* __restrict__ xich)
{
    const int i = blockIdx.x * 256 + threadIdx.x;       // over 4096*1024
    const int d = i & (DINNERX - 1);
    const int m = i >> 10;
    const int l = m & (LLX - 1);
    const int b = m >> 11;
    float acc = cb[d];
#pragma unroll
    for (int j = 0; j < DCONVX; ++j) {
        const int ls = l - (DCONVX - 1) + j;
        if (ls >= 0)
            acc += cw[d * DCONVX + j] * xz[((size_t)(b * LLX + ls) << 11) + d];
    }
    const float s = acc / (1.0f + __expf(-acc));        // SiLU
    xic[i]  = s;
    xich[i] = (_Float16)s;
}

// ---------------------------------------------------------------
// Selective scan: thread = (batch, channel); h[16] in VGPRs.
// Double-buffered LDS broadcast of B/C (one barrier/step), software-
// pipelined dt/xi/z loads. y -> f16 for the out_w GEMM.
// ---------------------------------------------------------------
__global__ __launch_bounds__(256)
void scan_kernel(const float* __restrict__ dtp,   // (M,1024) dt pre-softplus (+bias)
                 const float* __restrict__ xdbl,  // (M,64): cols 32..47=B, 48..63=C
                 const float* __restrict__ xz,    // (M,2048): z in cols 1024..2047
                 const float* __restrict__ xic,   // (M,1024) silu(conv(xi))
                 const float* __restrict__ A_log, // (1024,16)
                 const float* __restrict__ Dp,    // (1024,)
                 _Float16* __restrict__ yh)       // (M,1024) f16 y out
{
    const int tid = threadIdx.x;
    const int b = blockIdx.x >> 2;                       // 8 blocks: 2 batches * 4
    const int d = (blockIdx.x & 3) * 256 + tid;
    __shared__ float shB[2][DSTATE], shC[2][DSTATE];

    float Av[DSTATE];
#pragma unroll
    for (int n = 0; n < DSTATE; ++n) Av[n] = -__expf(A_log[d * DSTATE + n]);
    const float Dd = Dp[d];

    float h[DSTATE];
#pragma unroll
    for (int n = 0; n < DSTATE; ++n) h[n] = 0.0f;

    const size_t base = (size_t)b * LLX;

    if (tid < DSTATE)          shB[0][tid]          = xdbl[base * 64 + DTRANKX + tid];
    else if (tid < 2 * DSTATE) shC[0][tid - DSTATE] = xdbl[base * 64 + DTRANKX + DSTATE + (tid - DSTATE)];
    float dtr = dtp[base * DINNERX + d];
    float xiv = xic[base * DINNERX + d];
    float zv  = xz[(base << 11) + DINNERX + d];
    __syncthreads();

    for (int l = 0; l < LLX; ++l) {
        const int cur = l & 1;
        float dtr_n = 0.0f, xiv_n = 0.0f, zv_n = 0.0f;
        if (l + 1 < LLX) {                               // next-step loads first
            const size_t rn = base + l + 1;
            if (tid < DSTATE)          shB[1 - cur][tid]          = xdbl[rn * 64 + DTRANKX + tid];
            else if (tid < 2 * DSTATE) shC[1 - cur][tid - DSTATE] = xdbl[rn * 64 + DTRANKX + DSTATE + (tid - DSTATE)];
            dtr_n = dtp[rn * DINNERX + d];
            xiv_n = xic[rn * DINNERX + d];
            zv_n  = xz[(rn << 11) + DINNERX + d];
        }

        const float dtv = (dtr > 20.0f) ? dtr : log1pf(__expf(dtr));   // softplus
        const float dbx = dtv * xiv;
        float acc = 0.0f;
#pragma unroll
        for (int n = 0; n < DSTATE; ++n) {
            const float dA = __expf(dtv * Av[n]);
            h[n] = dA * h[n] + dbx * shB[cur][n];
            acc += h[n] * shC[cur][n];
        }
        const float sz = zv / (1.0f + __expf(-zv));
        yh[(base + l) * DINNERX + d] = (_Float16)((acc + xiv * Dd) * sz);

        dtr = dtr_n; xiv = xiv_n; zv = zv_n;
        __syncthreads();
    }
}

// ---------------------------------------------------------------
// asum[b,l,:] = out_f[b,l,:] + out_b[b,L-1-l,:]  -> f16 for final GEMM
// ---------------------------------------------------------------
__global__ __launch_bounds__(256)
void combine_kernel(const float* __restrict__ of, const float* __restrict__ ob,
                    _Float16* __restrict__ ash)
{
    const int i = blockIdx.x * 256 + threadIdx.x;       // over 4096*512
    const int c = i & (DIMX - 1);
    const int m = i >> 9;
    const int l = m & (LLX - 1);
    const int b = m >> 11;
    ash[i] = (_Float16)(of[i] + ob[((size_t)(b * LLX + (LLX - 1 - l)) << 9) + c]);
}

extern "C" void kernel_launch(void* const* d_in, const int* in_sizes, int n_in,
                              void* d_out, int out_size, void* d_ws, size_t ws_size,
                              hipStream_t stream)
{
    (void)in_sizes; (void)n_in; (void)out_size; (void)ws_size;

    const float* x      = (const float*)d_in[0];
    const float* ln_g   = (const float*)d_in[1];
    const float* ln_b   = (const float*)d_in[2];
    const float* proj_w = (const float*)d_in[3];
    const float* proj_b = (const float*)d_in[4];

    // ---- workspace carve-up: fp32 region then f16 region (~116 MB) ----
    float* ws   = (float*)d_ws;
    float* xz   = ws;                         // 4096*2048
    float* xic  = xz   + 8388608;             // 4096*1024
    float* xdbl = xic  + 4194304;             // 4096*64
    float* dtp  = xdbl + 262144;              // 4096*1024
    float* outf = dtp  + 4194304;             // 4096*512
    float* outb = outf + 2097152;             // 4096*512

    _Float16* hb      = (_Float16*)(outb + 2097152);
    _Float16* xnh     = hb;                   // 4096*512
    _Float16* xnrh    = xnh   + 2097152;      // 4096*512 (time-flipped)
    _Float16* xich    = xnrh  + 2097152;      // 4096*1024
    _Float16* xdblh   = xich  + 4194304;      // 4096*64
    _Float16* yh      = xdblh + 262144;       // 4096*1024
    _Float16* asumh   = yh    + 4194304;      // 4096*512
    _Float16* wt_in[2]  = { asumh + 2097152,           asumh + 2097152 + 1048576 };
    _Float16* wt_xp[2]  = { wt_in[1] + 1048576,        wt_in[1] + 1048576 + 65536 };
    _Float16* wt_dt[2]  = { wt_xp[1] + 65536,          wt_xp[1] + 65536 + 32768 };
    _Float16* wt_out[2] = { wt_dt[1] + 32768,          wt_dt[1] + 32768 + 524288 };
    _Float16* wt_proj   = wt_out[1] + 524288;
    float* out = (float*)d_out;

    // ---- one-time weight transpose+convert to f16 [N][K] ----
    wtrans_kernel<<<(512 * 512 + 255) / 256, 256, 0, stream>>>(proj_w, wt_proj, 512, 512);
    for (int dir = 0; dir < 2; ++dir) {
        wtrans_kernel<<<(512 * 2048 + 255) / 256, 256, 0, stream>>>(
            (const float*)d_in[5 + dir * 9 + 0], wt_in[dir], 512, 2048);
        wtrans_kernel<<<(1024 * 64 + 255) / 256, 256, 0, stream>>>(
            (const float*)d_in[5 + dir * 9 + 3], wt_xp[dir], 1024, XPROJN);
        wtrans_kernel<<<(32 * 1024 + 255) / 256, 256, 0, stream>>>(
            (const float*)d_in[5 + dir * 9 + 4], wt_dt[dir], DTRANKX, 1024);
        wtrans_kernel<<<(1024 * 512 + 255) / 256, 256, 0, stream>>>(
            (const float*)d_in[5 + dir * 9 + 8], wt_out[dir], 1024, 512);
    }

    layernorm_kernel<<<MROWS / 8, 256, 0, stream>>>(x, ln_g, ln_b, xnh, xnrh);

    for (int dir = 0; dir < 2; ++dir) {
        const float* conv_w = (const float*)d_in[5 + dir * 9 + 1];
        const float* conv_b = (const float*)d_in[5 + dir * 9 + 2];
        const float* dt_b   = (const float*)d_in[5 + dir * 9 + 5];
        const float* A_log  = (const float*)d_in[5 + dir * 9 + 6];
        const float* Dp     = (const float*)d_in[5 + dir * 9 + 7];

        // xz = (dir ? flip(xn) : xn) @ in_w        M=4096 N=2048 K=512
        gemm_wmma<128><<<dim3(MROWS / 64, 2048 / 128), 256, 0, stream>>>(
            dir ? xnrh : xnh, wt_in[dir], xz, MROWS, 2048, 512, 512, 2048,
            nullptr, nullptr, nullptr);

        conv_silu_kernel<<<(MROWS * DINNERX) / 256, 256, 0, stream>>>(
            xz, conv_w, conv_b, xic, xich);

        // xdbl = xic @ xproj_w (fp32 + f16 copy)   M=4096 N=64 K=1024
        gemm_wmma<64><<<dim3(MROWS / 64, XPROJN / 64), 256, 0, stream>>>(
            xich, wt_xp[dir], xdbl, MROWS, XPROJN, 1024, 1024, XPROJN,
            nullptr, nullptr, xdblh);

        // dtp = xdbl[:, :32] @ dt_w + dt_b         M=4096 N=1024 K=32
        gemm_wmma<128><<<dim3(MROWS / 64, DINNERX / 128), 256, 0, stream>>>(
            xdblh, wt_dt[dir], dtp, MROWS, DINNERX, DTRANKX, XPROJN, DINNERX,
            dt_b, nullptr, nullptr);

        scan_kernel<<<BBX * (DINNERX / 256), 256, 0, stream>>>(
            dtp, xdbl, xz, xic, A_log, Dp, yh);

        // out_dir = y @ out_w                      M=4096 N=512 K=1024
        gemm_wmma<128><<<dim3(MROWS / 64, DIMX / 128), 256, 0, stream>>>(
            yh, wt_out[dir], dir ? outb : outf, MROWS, DIMX, 1024, 1024, DIMX,
            nullptr, nullptr, nullptr);
    }

    combine_kernel<<<(MROWS * DIMX) / 256, 256, 0, stream>>>(outf, outb, asumh);

    // out = asum @ proj_w + proj_b + x             M=4096 N=512 K=512
    gemm_wmma<128><<<dim3(MROWS / 64, DIMX / 128), 256, 0, stream>>>(
        asumh, wt_proj, out, MROWS, DIMX, 512, 512, DIMX,
        proj_b, x, nullptr);
}
